// LinearAttention_36232344109196
// MI455X (gfx1250) — matrix operand order repaired and checked by reading, not compiled, not run on hardware
//
#include <hip/hip_runtime.h>
#include <math.h>

// ---------------- CDNA5 WMMA types ----------------
typedef __attribute__((ext_vector_type(16))) __bf16 v16bf;
typedef __attribute__((ext_vector_type(8)))  float  v8f;

union FragAB { v16bf v; unsigned int u[8]; };
union FragCD { v8f  v; float f[8]; };

// ---------------- problem constants ----------------
#define NB      8
#define HW      128        // image H == W
#define NPB     16384      // N per batch (HW*HW)
#define CCH     128        // channels
#define NHEADS  4
#define DH      32
#define ROWS    128        // rows (tokens) per workgroup == one image row
#define CHUNKS  128        // NPB / ROWS
#define NGROUPS 1024       // NB * CHUNKS
#define KT_STRIDE 136      // padded LDS stride (bf16 elems): 272B rows -> conflict free, 16B aligned
#define Q_STRIDE  132      // padded LDS stride for q

static __device__ inline unsigned int pk_bf16(float a, float b) {
  union { __bf16 h[2]; unsigned int u; } t;
  t.h[0] = (__bf16)a; t.h[1] = (__bf16)b;
  return t.u;
}
static __device__ inline float elu1(float v) { return v > 0.f ? v + 1.f : __expf(v); }

// =====================================================================
// Kernel 1: k-projection (WMMA bf16) + per-group kv / ksum partials
// =====================================================================
__global__ __launch_bounds__(256) void lin_attn_k1(
    const float* __restrict__ x, const float* __restrict__ Wqk,
    const float* __restrict__ bqk, float* __restrict__ kvpart,
    float* __restrict__ ksumpart) {
  __shared__ alignas(16) __bf16 kT[CCH * KT_STRIDE];   // k transposed: [channel][row]

  const int g    = blockIdx.x;          // g = b*CHUNKS + chunk
  const int tid  = threadIdx.x;
  const int wv   = tid >> 5;
  const int lane = tid & 31;
  const int l15  = lane & 15;
  const int hf   = lane >> 4;
  const float* xg = x + (size_t)g * ROWS * CCH;

  // ---- projection: k = elu(x @ Wk^T + bk) + 1 ; Wk = Wqk rows [128,256) ----
  {
    const int mt   = wv;               // 8 waves cover 8 M-tiles of 16 rows
    const int arow = mt * 16 + l15;
    FragAB Af[4];
    #pragma unroll
    for (int kc = 0; kc < 4; ++kc) {
      #pragma unroll
      for (int j = 0; j < 8; ++j) {    // A 16x32 bf16 layout (contiguous k = 16*hf + s)
        int c = kc * 32 + hf * 16 + 2 * j;
        Af[kc].u[j] = pk_bf16(xg[arow * CCH + c], xg[arow * CCH + c + 1]);
      }
    }
    for (int nt = 0; nt < 8; ++nt) {
      FragCD acc;
      #pragma unroll
      for (int r = 0; r < 8; ++r) acc.f[r] = 0.f;
      const int ncol = nt * 16 + l15;
      #pragma unroll
      for (int kc = 0; kc < 4; ++kc) {
        FragAB Bf;                     // B 32x16 bf16: lane=col, K pairs along regs
        const float* wr = Wqk + (size_t)(CCH + ncol) * CCH + kc * 32 + hf * 16;
        #pragma unroll
        for (int j = 0; j < 8; ++j) Bf.u[j] = pk_bf16(wr[2 * j], wr[2 * j + 1]);
        acc.v = __builtin_amdgcn_wmma_f32_16x16x32_bf16(
            false, Af[kc].v, false, Bf.v, (short)0, acc.v, false, false);
      }
      const float bias    = bqk[CCH + ncol];
      const int   rowbase = mt * 16 + hf * 8;   // D reg r -> row M = r + 8*hf
      unsigned int pw[4];
      #pragma unroll
      for (int j = 0; j < 4; ++j)
        pw[j] = pk_bf16(elu1(acc.f[2 * j] + bias), elu1(acc.f[2 * j + 1] + bias));
      uint4 p; p.x = pw[0]; p.y = pw[1]; p.z = pw[2]; p.w = pw[3];
      *(uint4*)&kT[ncol * KT_STRIDE + rowbase] = p;   // 16B ds_store, transposed
    }
  }
  __syncthreads();

  // ---- kv partial: kv[d][e] = sum_rows k[row][d] * v[row][e]  (v = x) ----
  {
    const int h = wv >> 1, mtq = wv & 1;
    FragCD acc[2];
    #pragma unroll
    for (int q = 0; q < 2; ++q)
      #pragma unroll
      for (int r = 0; r < 8; ++r) acc[q].f[r] = 0.f;
    const int drow = h * DH + mtq * 16 + l15;   // k-channel (A matrix row m)
    for (int kc = 0; kc < 4; ++kc) {
      FragAB Af;                                 // A = k^T (m=d, k=row): contiguous in kT
      #pragma unroll
      for (int j = 0; j < 8; ++j) {
        int rr = kc * 32 + hf * 16 + 2 * j;
        Af.u[j] = *(const unsigned int*)&kT[drow * KT_STRIDE + rr];
      }
      #pragma unroll
      for (int nt = 0; nt < 2; ++nt) {
        FragAB Bf;                               // B = v (k=row, n=e): gather from x (L2)
        const int chan = h * DH + nt * 16 + l15;
        #pragma unroll
        for (int j = 0; j < 8; ++j) {
          int rr = kc * 32 + hf * 16 + 2 * j;
          Bf.u[j] = pk_bf16(xg[rr * CCH + chan], xg[(rr + 1) * CCH + chan]);
        }
        acc[nt].v = __builtin_amdgcn_wmma_f32_16x16x32_bf16(
            false, Af.v, false, Bf.v, (short)0, acc[nt].v, false, false);
      }
    }
    float* kvp = kvpart + ((size_t)g * NHEADS + h) * (DH * DH);
    #pragma unroll
    for (int nt = 0; nt < 2; ++nt)
      #pragma unroll
      for (int r = 0; r < 8; ++r) {
        int dl = mtq * 16 + r + 8 * hf;
        int el = nt * 16 + l15;
        kvp[dl * DH + el] = acc[nt].f[r];
      }
  }

  // ---- ksum partial (per channel over 128 rows) ----
  if (tid < CCH) {
    float s = 0.f;
    for (int r = 0; r < ROWS; ++r) s += (float)kT[tid * KT_STRIDE + r];
    ksumpart[(size_t)g * CCH + tid] = s;
  }
}

// =====================================================================
// Kernel 2: reduce partials -> kv (scaled 1/N) and k_mean
// =====================================================================
__global__ __launch_bounds__(256) void lin_attn_k2(
    const float* __restrict__ kvpart, const float* __restrict__ ksumpart,
    float* __restrict__ kvf, float* __restrict__ kmean) {
  const int t = blockIdx.x * 256 + threadIdx.x;
  const float invN = 1.f / (float)NPB;
  if (t < NB * NHEADS * DH * DH) {                 // 32768 kv outputs
    const int b = t >> 12, idx = t & 4095;
    const float* p = kvpart + (size_t)b * CHUNKS * 4096 + idx;
    float s = 0.f;
    for (int ch = 0; ch < CHUNKS; ++ch) s += p[(size_t)ch * 4096];
    kvf[t] = s * invN;
  } else if (t < NB * NHEADS * DH * DH + NB * CCH) {  // 1024 kmean outputs
    const int u = t - NB * NHEADS * DH * DH;
    const int b = u >> 7, c = u & 127;
    const float* p = ksumpart + (size_t)b * CHUNKS * CCH + c;
    float s = 0.f;
    for (int ch = 0; ch < CHUNKS; ++ch) s += p[(size_t)ch * CCH];
    kmean[u] = s * invN;
  }
}

// =====================================================================
// Kernel 3: q-projection (WMMA) + z + out^T = kv^T * q^T (WMMA) + LePE conv
// =====================================================================
__global__ __launch_bounds__(256) void lin_attn_k3(
    const float* __restrict__ x, const float* __restrict__ Wqk,
    const float* __restrict__ bqk, const float* __restrict__ lepe_w,
    const float* __restrict__ lepe_b, const float* __restrict__ kvf,
    const float* __restrict__ kmean, float* __restrict__ out) {
  __shared__ alignas(16) __bf16 qs[ROWS * Q_STRIDE]; // q bf16, row-major
  __shared__ float zs[ROWS * NHEADS];
  __shared__ float kms[CCH];
  __shared__ float w9[CCH * 9];
  __shared__ float wb[CCH];

  const int g    = blockIdx.x;           // one image row per group
  const int b    = g >> 7;
  const int yrow = g & 127;
  const int tid  = threadIdx.x;
  const int wv   = tid >> 5;
  const int lane = tid & 31;
  const int l15  = lane & 15;
  const int hf   = lane >> 4;
  const float* xg = x + (size_t)g * ROWS * CCH;

  for (int i = tid; i < CCH * 9; i += 256) w9[i] = lepe_w[i];
  if (tid < CCH) { wb[tid] = lepe_b[tid]; kms[tid] = kmean[b * CCH + tid]; }

  // ---- projection: q = elu(x @ Wq^T + bq) + 1 ; Wq = Wqk rows [0,128) ----
  {
    const int mt   = wv;
    const int arow = mt * 16 + l15;
    FragAB Af[4];
    #pragma unroll
    for (int kc = 0; kc < 4; ++kc) {
      #pragma unroll
      for (int j = 0; j < 8; ++j) {
        int c = kc * 32 + hf * 16 + 2 * j;
        Af[kc].u[j] = pk_bf16(xg[arow * CCH + c], xg[arow * CCH + c + 1]);
      }
    }
    for (int nt = 0; nt < 8; ++nt) {
      FragCD acc;
      #pragma unroll
      for (int r = 0; r < 8; ++r) acc.f[r] = 0.f;
      const int ncol = nt * 16 + l15;
      #pragma unroll
      for (int kc = 0; kc < 4; ++kc) {
        FragAB Bf;
        const float* wr = Wqk + (size_t)ncol * CCH + kc * 32 + hf * 16;
        #pragma unroll
        for (int j = 0; j < 8; ++j) Bf.u[j] = pk_bf16(wr[2 * j], wr[2 * j + 1]);
        acc.v = __builtin_amdgcn_wmma_f32_16x16x32_bf16(
            false, Af[kc].v, false, Bf.v, (short)0, acc.v, false, false);
      }
      const float bias    = bqk[ncol];
      const int   rowbase = mt * 16 + hf * 8;
      #pragma unroll
      for (int r = 0; r < 8; ++r)
        qs[(rowbase + r) * Q_STRIDE + ncol] = (__bf16)elu1(acc.f[r] + bias);
    }
  }
  __syncthreads();

  // ---- z = 1 / (q . k_mean + eps) per (row, head) ----
  for (int t = tid; t < ROWS * NHEADS; t += 256) {
    const int row = t >> 2, h = t & 3;
    float s = 0.f;
    for (int d = 0; d < DH; ++d)
      s += (float)qs[row * Q_STRIDE + h * DH + d] * kms[h * DH + d];
    zs[t] = 1.f / (s + 1e-6f);
  }
  __syncthreads();

  // ---- out^T = kv^T * q^T : D lane-column == output row -> per-lane z, fused LePE ----
  {
    const int h = wv >> 1, mtq = wv & 1;
    const float* kvh = kvf + ((size_t)b * NHEADS + h) * (DH * DH);
    FragAB Af;                                  // A = kv^T (m=e, k=d)
    const int e = mtq * 16 + l15;
    #pragma unroll
    for (int j = 0; j < 8; ++j) {
      int d = hf * 16 + 2 * j;
      Af.u[j] = pk_bf16(kvh[d * DH + e], kvh[(d + 1) * DH + e]);
    }
    for (int ntr = 0; ntr < 8; ++ntr) {
      const int row = ntr * 16 + l15;           // local row == xw position
      FragAB Bf;                                // B = q^T (k=d, n=row): contiguous in qs
      #pragma unroll
      for (int j = 0; j < 8; ++j) {
        int d = hf * 16 + 2 * j;
        Bf.u[j] = *(const unsigned int*)&qs[row * Q_STRIDE + h * DH + d];
      }
      FragCD acc;
      #pragma unroll
      for (int r = 0; r < 8; ++r) acc.f[r] = 0.f;
      acc.v = __builtin_amdgcn_wmma_f32_16x16x32_bf16(
          false, Af.v, false, Bf.v, (short)0, acc.v, false, false);

      const float z  = zs[row * NHEADS + h];
      const int   cb = h * DH + mtq * 16 + 8 * hf;  // 8 consecutive channels per lane
      float lep[8];
      #pragma unroll
      for (int r = 0; r < 8; ++r) lep[r] = wb[cb + r];
      // depthwise 3x3 "SAME" conv on v (= x), reads served by L2 (x resident)
      #pragma unroll
      for (int ky = 0; ky < 3; ++ky) {
        const int yy = yrow + ky - 1;
        if (yy < 0 || yy >= HW) continue;
        #pragma unroll
        for (int kx = 0; kx < 3; ++kx) {
          const int xx = row + kx - 1;
          if (xx < 0 || xx >= HW) continue;
          const float* src = x + (((size_t)b * NPB) + (size_t)yy * HW + xx) * CCH + cb;
          const float4 lo = *(const float4*)src;
          const float4 hi = *(const float4*)(src + 4);
          const int k9 = ky * 3 + kx;
          lep[0] += lo.x * w9[(cb + 0) * 9 + k9];
          lep[1] += lo.y * w9[(cb + 1) * 9 + k9];
          lep[2] += lo.z * w9[(cb + 2) * 9 + k9];
          lep[3] += lo.w * w9[(cb + 3) * 9 + k9];
          lep[4] += hi.x * w9[(cb + 4) * 9 + k9];
          lep[5] += hi.y * w9[(cb + 5) * 9 + k9];
          lep[6] += hi.z * w9[(cb + 6) * 9 + k9];
          lep[7] += hi.w * w9[(cb + 7) * 9 + k9];
        }
      }
      float* op = out + ((size_t)g * ROWS + row) * CCH + cb;
      float4 o0, o1;
      o0.x = acc.f[0] * z + lep[0];  o0.y = acc.f[1] * z + lep[1];
      o0.z = acc.f[2] * z + lep[2];  o0.w = acc.f[3] * z + lep[3];
      o1.x = acc.f[4] * z + lep[4];  o1.y = acc.f[5] * z + lep[5];
      o1.z = acc.f[6] * z + lep[6];  o1.w = acc.f[7] * z + lep[7];
      *(float4*)op       = o0;
      *(float4*)(op + 4) = o1;
    }
  }
}

// =====================================================================
extern "C" void kernel_launch(void* const* d_in, const int* in_sizes, int n_in,
                              void* d_out, int out_size, void* d_ws, size_t ws_size,
                              hipStream_t stream) {
  (void)in_sizes; (void)n_in; (void)out_size; (void)ws_size;
  const float* x      = (const float*)d_in[0];
  const float* Wqk    = (const float*)d_in[1];
  const float* bqk    = (const float*)d_in[2];
  const float* lepe_w = (const float*)d_in[3];
  const float* lepe_b = (const float*)d_in[4];
  float* out = (float*)d_out;

  // workspace layout (floats), ~17.4 MB total; everything written before read
  float* ws       = (float*)d_ws;
  float* kvpart   = ws;                                                  // NGROUPS*4096
  float* ksumpart = kvpart + (size_t)NGROUPS * NHEADS * DH * DH;         // NGROUPS*128
  float* kvf      = ksumpart + (size_t)NGROUPS * CCH;                    // 8*4096
  float* kmean    = kvf + (size_t)NB * NHEADS * DH * DH;                 // 8*128

  lin_attn_k1<<<NGROUPS, 256, 0, stream>>>(x, Wqk, bqk, kvpart, ksumpart);
  lin_attn_k2<<<(NB * NHEADS * DH * DH + NB * CCH + 255) / 256, 256, 0, stream>>>(
      kvpart, ksumpart, kvf, kmean);
  lin_attn_k3<<<NGROUPS, 256, 0, stream>>>(x, Wqk, bqk, lepe_w, lepe_b, kvf, kmean, out);
}
